// Model_49795850830401
// MI455X (gfx1250) — matrix operand order repaired and checked by reading, not compiled, not run on hardware
//
#include <hip/hip_runtime.h>
#include <cmath>

// ---------- CDNA5 WMMA types ----------
typedef __attribute__((ext_vector_type(16))) __bf16 v16bf;
typedef __attribute__((ext_vector_type(8)))  float  v8f;

#define BATCH    4096
#define NSTEPS   64
#define DIMX     100
#define HDIM     512
#define XSTRIDE  128           // padded x-buffer (100 cols used, 100..127 == 0)
#define K1PAD    128           // padded K for layer 1
#define W4NPAD   128           // padded N for layer 4 (100 real cols)
#define BN_EPS   1e-5f
#define INV_B    (1.0f / 4096.0f)
#define SQRT2F   1.41421356237f
#define LDA      56            // LDS row stride (112B = 7*16B, conflict-friendly)

// ---------- helpers ----------
__device__ __forceinline__ float gelu_f(float x) {
    return 0.5f * x * (1.0f + erff(x * 0.70710678118f));
}

__device__ __forceinline__ unsigned pcg_hash(unsigned v) {
    v = v * 747796405u + 2891336453u;
    v = ((v >> ((v >> 28) + 4u)) ^ v) * 277803737u;
    return (v >> 22) ^ v;
}

// one standard normal per (step,row,d); deterministic counter-based
__device__ __forceinline__ float gauss1(unsigned step, unsigned row, unsigned d) {
    unsigned s  = step * 0x9E3779B9u ^ (row * 0x85EBCA77u + 0x165667B1u) ^ (d * 0x27D4EB2Fu);
    unsigned h0 = pcg_hash(s);
    unsigned h1 = pcg_hash(h0 ^ 0x68E31DA4u);
    float u1 = ((h0 >> 8) + 1u) * (1.0f / 16777216.0f);   // (0,1]
    float u2 = (h1 >> 8) * (1.0f / 16777216.0f);          // [0,1)
    float r  = sqrtf(-2.0f * logf(u1));
    return r * cosf(6.28318530718f * u2);
}

// ---------- CDNA5 async global->LDS copies (ASYNCcnt path) ----------
__device__ __forceinline__ unsigned lds_off(const void* p) {
    return (unsigned)(unsigned long long)(uintptr_t)p;   // low 32 bits = LDS byte addr
}
__device__ __forceinline__ void async_b128(unsigned lds, const __bf16* g) {
    asm volatile("global_load_async_to_lds_b128 %0, %1, off"
                 :: "v"(lds), "v"(g) : "memory");
}
__device__ __forceinline__ void async_b64(unsigned lds, const __bf16* g) {
    asm volatile("global_load_async_to_lds_b64 %0, %1, off"
                 :: "v"(lds), "v"(g) : "memory");
}
__device__ __forceinline__ void wait_async3() {
    asm volatile("s_wait_asynccnt 0x3" ::: "memory");
}
__device__ __forceinline__ void wait_async0() {
    asm volatile("s_wait_asynccnt 0x0" ::: "memory");
}

// A fragment, 16x32 bf16: elems 0..7 = K khalf..+7, 8..15 = K khalf+16..+23
__device__ __forceinline__ v16bf load_frag_a(const __bf16* As, int m, int lane) {
    int khalf = (lane >> 4) * 8;
    union { v16bf v; uint4 q[2]; } f;
    const __bf16* p = As + m * LDA;
    f.q[0] = *reinterpret_cast<const uint4*>(p + khalf);
    f.q[1] = *reinterpret_cast<const uint4*>(p + khalf + 16);
    return f.v;
}
// B fragment, 32x16 bf16 (stored [n][k]): elems 0..7 = K kbase..+7, 8..15 = K kbase+16..+23, kbase = (lane>=16)?8:0
__device__ __forceinline__ v16bf load_frag_b(const __bf16* Bs, int n, int lane) {
    int kbase = (lane >> 4) * 8;
    union { v16bf v; uint4 q[2]; } f;
    const __bf16* p = Bs + n * LDA + kbase;
    f.q[0] = *reinterpret_cast<const uint4*>(p);
    f.q[1] = *reinterpret_cast<const uint4*>(p + 16);
    return f.v;
}

// ---------- prep kernels ----------
__global__ void cvt_wt_k(const float* __restrict__ W, __bf16* __restrict__ Wt,
                         int Ksrc, int Kpad, int Nsrc, int Npad) {
    int idx = blockIdx.x * blockDim.x + threadIdx.x;
    if (idx >= Npad * Kpad) return;
    int n = idx / Kpad, k = idx % Kpad;
    float v = (k < Ksrc && n < Nsrc) ? W[(size_t)k * Nsrc + n] : 0.0f;
    Wt[idx] = (__bf16)v;
}

__global__ void init_state_k(const float* __restrict__ x0, const float* __restrict__ y0,
                             float* __restrict__ xbuf, __bf16* __restrict__ xbf,
                             float* __restrict__ ybuf) {
    int idx = blockIdx.x * blockDim.x + threadIdx.x;
    if (idx >= BATCH * XSTRIDE) return;
    int c = idx & (XSTRIDE - 1);
    float v = (c < DIMX) ? x0[c] : 0.0f;
    xbuf[idx] = v;
    xbf[idx]  = (__bf16)v;
    if (c == 0) ybuf[idx >> 7] = y0[0];
}

__global__ void zero_stats_k(float* __restrict__ stats) {
    for (int i = threadIdx.x; i < 4 * HDIM; i += blockDim.x) stats[i] = 0.0f;
}

// Fold BatchNorm (per-channel affine a_k*h+c_k) into next layer's weights:
// Ws[n][k] = a_k*Wt[n][k];  bias_out[n] = bias_in[n] + sum_k c_k*Wt[n][k]
__global__ __launch_bounds__(256)
void fold_bn_k(const __bf16* __restrict__ wt, __bf16* __restrict__ ws,
               const float* __restrict__ sum, const float* __restrict__ sq,
               const float* __restrict__ g, const float* __restrict__ be,
               const float* __restrict__ bias_in, int bias_n,
               float* __restrict__ bias_out, int kdim) {
    __shared__ float red[256];
    const int n = blockIdx.x;
    float acc = 0.0f;
    for (int k = threadIdx.x; k < kdim; k += 256) {
        float m   = sum[k] * INV_B;
        float var = sq[k] * INV_B - m * m;
        float a   = g[k] * rsqrtf(var + BN_EPS);
        float c   = be[k] - a * m;
        float w   = (float)wt[(size_t)n * kdim + k];
        ws[(size_t)n * kdim + k] = (__bf16)(a * w);
        acc += c * w;
    }
    red[threadIdx.x] = acc;
    __syncthreads();
    for (int s = 128; s > 0; s >>= 1) {
        if (threadIdx.x < s) red[threadIdx.x] += red[threadIdx.x + s];
        __syncthreads();
    }
    if (threadIdx.x == 0)
        bias_out[n] = ((n < bias_n) ? bias_in[n] : 0.0f) + red[0];
}

__global__ void writeout_k(const float* __restrict__ xbuf, const float* __restrict__ ybuf,
                           float* __restrict__ out) {
    int idx = blockIdx.x * blockDim.x + threadIdx.x;
    if (idx < BATCH * DIMX) {
        out[idx] = xbuf[(idx / DIMX) * XSTRIDE + (idx % DIMX)];
    } else if (idx < BATCH * DIMX + BATCH) {
        out[idx] = ybuf[idx - BATCH * DIMX];
    }
}

// ---------- pure bf16 GEMM + bias + GELU (+optional column stats) ----------
// out[4096,512] = gelu(A[4096,K] @ Wt^T + bias); Wt stored [512][K] bf16.
// 256 thr = 8 waves; block tile 128x64; wave tile 32x32 (4 wmma / 32-K chunk).
// Async double-buffered LDS pipeline: 3 async ops per wave per chunk.
template <bool STATS, bool TBIAS>
__global__ __launch_bounds__(256)
void gemm_k(const __bf16* __restrict__ inp, const __bf16* __restrict__ wt, int kdim,
            const float* __restrict__ bias, const float* __restrict__ w1last, float tval,
            float* __restrict__ st_sum, float* __restrict__ st_sq,
            __bf16* __restrict__ out) {
    __shared__ __align__(16) __bf16 As[2][128][LDA];
    __shared__ __align__(16) __bf16 Bs[2][64][LDA];

    const int rowBase = blockIdx.x * 128;
    const int colBase = blockIdx.y * 64;
    const int tx   = threadIdx.x;
    const int lane = tx & 31;
    const int wid  = tx >> 5;
    const int wm   = wid & 3;    // 4 M bands of 32
    const int wn   = wid >> 2;   // 2 N bands of 32

    const unsigned baseA = lds_off(&As[0][0][0]);
    const unsigned baseB = lds_off(&Bs[0][0][0]);

    v8f acc[2][2] = {{{0,0,0,0,0,0,0,0},{0,0,0,0,0,0,0,0}},
                     {{0,0,0,0,0,0,0,0},{0,0,0,0,0,0,0,0}}};

    const int nc = kdim / 32;

    // ---- stage(chunk, buf): 2 A-b128 + 1 B-b128 per thread ----
#define STAGE(CH, BUF)                                                          \
    do {                                                                        \
        int kc = (CH) * 32;                                                     \
        _Pragma("unroll")                                                       \
        for (int j = 0; j < 2; ++j) {                                           \
            int s = tx + 256 * j;                                               \
            int r = s >> 2, kk = (s & 3) * 8;                                   \
            async_b128(baseA + (BUF) * (128 * 2 * LDA) + r * (2 * LDA) + kk * 2,\
                       inp + (size_t)(rowBase + r) * kdim + kc + kk);           \
        }                                                                       \
        {                                                                       \
            int r = tx >> 2, kk = (tx & 3) * 8;                                 \
            async_b128(baseB + (BUF) * (64 * 2 * LDA) + r * (2 * LDA) + kk * 2, \
                       wt + (size_t)(colBase + r) * kdim + kc + kk);            \
        }                                                                       \
    } while (0)

    STAGE(0, 0);
    STAGE(1, 1);

    for (int i = 0; i < nc; ++i) {
        if (i + 1 < nc) wait_async3(); else wait_async0();
        __syncthreads();

        const __bf16* Ab = &As[i & 1][0][0];
        const __bf16* Bb = &Bs[i & 1][0][0];
        v16bf a0 = load_frag_a(Ab, 32 * wm + (lane & 15), lane);
        v16bf a1 = load_frag_a(Ab, 32 * wm + 16 + (lane & 15), lane);
        v16bf b0 = load_frag_b(Bb, 32 * wn + (lane & 15), lane);
        v16bf b1 = load_frag_b(Bb, 32 * wn + 16 + (lane & 15), lane);
        acc[0][0] = __builtin_amdgcn_wmma_f32_16x16x32_bf16(false, a0, false, b0, (short)0, acc[0][0], false, false);
        acc[0][1] = __builtin_amdgcn_wmma_f32_16x16x32_bf16(false, a0, false, b1, (short)0, acc[0][1], false, false);
        acc[1][0] = __builtin_amdgcn_wmma_f32_16x16x32_bf16(false, a1, false, b0, (short)0, acc[1][0], false, false);
        acc[1][1] = __builtin_amdgcn_wmma_f32_16x16x32_bf16(false, a1, false, b1, (short)0, acc[1][1], false, false);

        __syncthreads();
        if (i + 2 < nc) STAGE(i + 2, i & 1);
    }
#undef STAGE

    // ---- epilogue: bias + GELU, store bf16, optional column stats ----
    const int rhalf = (lane >> 4) * 8;
#pragma unroll
    for (int tt = 0; tt < 2; ++tt) {
        int col = colBase + 32 * wn + 16 * tt + (lane & 15);
        float bv = bias[col];
        if (TBIAS) bv += tval * w1last[col];
        float s = 0.0f, s2 = 0.0f;
#pragma unroll
        for (int mt = 0; mt < 2; ++mt) {
#pragma unroll
            for (int vg = 0; vg < 8; ++vg) {
                int row = rowBase + 32 * wm + 16 * mt + rhalf + vg;
                float g = gelu_f(acc[mt][tt][vg] + bv);
                out[(size_t)row * HDIM + col] = (__bf16)g;
                s += g; s2 += g * g;
            }
        }
        if (STATS) {
            atomicAdd(&st_sum[col], s);
            atomicAdd(&st_sq[col], s2);
        }
    }
}

// ---------- tail: layer-4 GEMM (BN pre-folded) + SDE update ----------
// block = 32 rows x 128 cols; 8 waves: 2 M bands x 4 N bands (2 wmma/chunk).
__global__ __launch_bounds__(256)
void tail_k(const __bf16* __restrict__ h3, const __bf16* __restrict__ w4s,
            const float* __restrict__ bias4,
            float* __restrict__ xbuf, __bf16* __restrict__ xbf,
            float* __restrict__ ybuf, int step, float dt, float sqdt) {
    __shared__ __align__(16) __bf16 As[2][32][LDA];
    __shared__ __align__(16) __bf16 Bs[2][128][LDA];
    __shared__ float zs[32][136];
    __shared__ float part1[32][8];
    __shared__ float part2[32][8];

    const int rowBase = blockIdx.x * 32;
    const int tx   = threadIdx.x;
    const int lane = tx & 31;
    const int wid  = tx >> 5;
    const int wm   = wid & 1;    // 2 M bands of 16
    const int wn   = wid >> 1;   // 4 N bands of 32

    const unsigned baseA = lds_off(&As[0][0][0]);
    const unsigned baseB = lds_off(&Bs[0][0][0]);

    v8f acc[2] = {{0,0,0,0,0,0,0,0}, {0,0,0,0,0,0,0,0}};

#define STAGE_T(CH, BUF)                                                        \
    do {                                                                        \
        int kc = (CH) * 32;                                                     \
        {   /* A: 32x32, one b64 (4 bf16) per thread */                         \
            int r = tx >> 3, kk = (tx & 7) * 4;                                 \
            async_b64(baseA + (BUF) * (32 * 2 * LDA) + r * (2 * LDA) + kk * 2,  \
                      h3 + (size_t)(rowBase + r) * HDIM + kc + kk);             \
        }                                                                       \
        _Pragma("unroll")                                                       \
        for (int j = 0; j < 2; ++j) { /* B: 128x32, two b128 per thread */      \
            int s = tx + 256 * j;                                               \
            int r = s >> 2, kk = (s & 3) * 8;                                   \
            async_b128(baseB + (BUF) * (128 * 2 * LDA) + r * (2 * LDA) + kk * 2,\
                       w4s + (size_t)r * HDIM + kc + kk);                       \
        }                                                                       \
    } while (0)

    STAGE_T(0, 0);
    STAGE_T(1, 1);

    const int nc = HDIM / 32;
    for (int i = 0; i < nc; ++i) {
        if (i + 1 < nc) wait_async3(); else wait_async0();
        __syncthreads();

        const __bf16* Ab = &As[i & 1][0][0];
        const __bf16* Bb = &Bs[i & 1][0][0];
        v16bf av = load_frag_a(Ab, 16 * wm + (lane & 15), lane);
#pragma unroll
        for (int tt = 0; tt < 2; ++tt) {
            v16bf bv = load_frag_b(Bb, 32 * wn + 16 * tt + (lane & 15), lane);
            acc[tt] = __builtin_amdgcn_wmma_f32_16x16x32_bf16(
                false, av, false, bv, (short)0, acc[tt], false, false);
        }
        __syncthreads();
        if (i + 2 < nc) STAGE_T(i + 2, i & 1);
    }
#undef STAGE_T

    // z -> LDS (fp32)
    const int rhalf = (lane >> 4) * 8;
#pragma unroll
    for (int tt = 0; tt < 2; ++tt) {
        int c = 32 * wn + 16 * tt + (lane & 15);
        float bv = bias4[c];
#pragma unroll
        for (int vg = 0; vg < 8; ++vg) {
            int r = 16 * wm + rhalf + vg;
            zs[r][c] = acc[tt][vg] + bv;
        }
    }
    __syncthreads();

    // SDE update: 8 threads per row
    const int rl  = tx >> 3;
    const int sub = tx & 7;
    const unsigned grow = rowBase + rl;
    float zdw = 0.0f, z2 = 0.0f;
    for (int d = sub; d < DIMX; d += 8) {
        float z  = zs[rl][d];
        float dw = sqdt * gauss1((unsigned)step, grow, (unsigned)d);
        zdw += z * dw;
        z2  += z * z;
        size_t xi = (size_t)grow * XSTRIDE + d;
        float nx = xbuf[xi] + SQRT2F * dw;      // x += sqrt(2)*dW
        xbuf[xi] = nx;
        xbf[xi]  = (__bf16)nx;                  // bf16 mirror for next L1 GEMM
    }
    part1[rl][sub] = zdw;
    part2[rl][sub] = z2;
    __syncthreads();
    if (sub == 0) {
        float s1 = 0.0f, s2 = 0.0f;
#pragma unroll
        for (int i = 0; i < 8; ++i) { s1 += part1[rl][i]; s2 += part2[rl][i]; }
        ybuf[grow] += 0.5f * s2 * dt + s1;      // y = y - f*dt + z.dW, f = -0.5*||z||^2
    }
}

// ---------- launch ----------
extern "C" void kernel_launch(void* const* d_in, const int* in_sizes, int n_in,
                              void* d_out, int out_size, void* d_ws, size_t ws_size,
                              hipStream_t stream) {
    (void)in_sizes; (void)n_in; (void)out_size; (void)ws_size;
    const float* y0  = (const float*)d_in[0];
    const float* W1  = (const float*)d_in[1];
    const float* b1  = (const float*)d_in[2];
    const float* W2  = (const float*)d_in[3];
    const float* b2  = (const float*)d_in[4];
    const float* W3  = (const float*)d_in[5];
    const float* b3  = (const float*)d_in[6];
    const float* W4  = (const float*)d_in[7];
    const float* b4  = (const float*)d_in[8];
    const float* g1  = (const float*)d_in[9];
    const float* be1 = (const float*)d_in[10];
    const float* g2  = (const float*)d_in[11];
    const float* be2 = (const float*)d_in[12];
    const float* x0  = (const float*)d_in[13];
    // d_in[14]=batch_size(4096), d_in[15]=N(64): compile-time constants here.

    uint8_t* ws = (uint8_t*)d_ws;
    size_t off = 0;
    auto take = [&](size_t bytes) { uint8_t* p = ws + off; off = (off + bytes + 255) & ~(size_t)255; return p; };
    float*  xbuf  = (float*)take((size_t)BATCH * XSTRIDE * 4);
    __bf16* xbf   = (__bf16*)take((size_t)BATCH * XSTRIDE * 2);
    float*  ybuf  = (float*)take((size_t)BATCH * 4);
    __bf16* h1    = (__bf16*)take((size_t)BATCH * HDIM * 2);
    __bf16* h2    = (__bf16*)take((size_t)BATCH * HDIM * 2);
    __bf16* h3    = (__bf16*)take((size_t)BATCH * HDIM * 2);
    __bf16* w1t   = (__bf16*)take((size_t)HDIM * K1PAD * 2);
    __bf16* w2t   = (__bf16*)take((size_t)HDIM * HDIM * 2);
    __bf16* w3t   = (__bf16*)take((size_t)HDIM * HDIM * 2);
    __bf16* w3s   = (__bf16*)take((size_t)HDIM * HDIM * 2);
    __bf16* w4t   = (__bf16*)take((size_t)W4NPAD * HDIM * 2);
    __bf16* w4s   = (__bf16*)take((size_t)W4NPAD * HDIM * 2);
    float*  stats = (float*)take(4 * HDIM * 4);
    float*  b3e   = (float*)take(HDIM * 4);
    float*  b4e   = (float*)take(W4NPAD * 4);
    float* sum1 = stats, *sq1 = stats + HDIM, *sum2 = stats + 2 * HDIM, *sq2 = stats + 3 * HDIM;

    // ---- prep (re-run every call; deterministic) ----
    cvt_wt_k<<<(HDIM * K1PAD + 255) / 256, 256, 0, stream>>>(W1, w1t, DIMX, K1PAD, HDIM, HDIM);
    cvt_wt_k<<<(HDIM * HDIM + 255) / 256, 256, 0, stream>>>(W2, w2t, HDIM, HDIM, HDIM, HDIM);
    cvt_wt_k<<<(HDIM * HDIM + 255) / 256, 256, 0, stream>>>(W3, w3t, HDIM, HDIM, HDIM, HDIM);
    cvt_wt_k<<<(W4NPAD * HDIM + 255) / 256, 256, 0, stream>>>(W4, w4t, HDIM, HDIM, DIMX, W4NPAD);
    init_state_k<<<(BATCH * XSTRIDE + 255) / 256, 256, 0, stream>>>(x0, y0, xbuf, xbf, ybuf);

    const float dt   = 1.0f / (float)NSTEPS;
    const float sqdt = sqrtf(dt);
    const dim3 gFull(BATCH / 128, HDIM / 64);   // (32, 8)

    for (int s = 0; s < NSTEPS; ++s) {
        float t = (float)s * dt;
        zero_stats_k<<<1, 256, 0, stream>>>(stats);
        // L1: bf16 x in (padded K=128), t*W1[100,:] folded into bias
        gemm_k<false, true><<<gFull, 256, 0, stream>>>(
            xbf, w1t, K1PAD, b1, W1 + (size_t)DIMX * HDIM, t, nullptr, nullptr, h1);
        // L2: + accumulate BN1 stats (post-GELU)
        gemm_k<true, false><<<gFull, 256, 0, stream>>>(
            h1, w2t, HDIM, b2, nullptr, 0.0f, sum1, sq1, h2);
        // fold BN1 into W3
        fold_bn_k<<<HDIM, 256, 0, stream>>>(w3t, w3s, sum1, sq1, g1, be1, b3, HDIM, b3e, HDIM);
        // L3: + accumulate BN2 stats
        gemm_k<true, false><<<gFull, 256, 0, stream>>>(
            h2, w3s, HDIM, b3e, nullptr, 0.0f, sum2, sq2, h3);
        // fold BN2 into W4
        fold_bn_k<<<W4NPAD, 256, 0, stream>>>(w4t, w4s, sum2, sq2, g2, be2, b4, DIMX, b4e, HDIM);
        // L4 + SDE update
        tail_k<<<BATCH / 32, 256, 0, stream>>>(
            h3, w4s, b4e, xbuf, xbf, ybuf, s, dt, sqdt);
    }

    writeout_k<<<(BATCH * DIMX + BATCH + 255) / 256, 256, 0, stream>>>(
        xbuf, ybuf, (float*)d_out);
}
